// LSTMscratch_47596827574336
// MI455X (gfx1250) — compile-verified, matrix operand order checked by reading, not run on hardware
//
#include <hip/hip_runtime.h>
#include <hip/hip_bf16.h>

// ---------- types ----------
typedef __attribute__((ext_vector_type(16))) __bf16 vbf16;
typedef __attribute__((ext_vector_type(8)))  float  v8f;
typedef __attribute__((ext_vector_type(8)))  unsigned short us8;

// ---------- helpers ----------
static __device__ __forceinline__ unsigned short f2b(float f) {
    union { float f; unsigned u; } v; v.f = f;
    unsigned r = v.u + 0x7FFFu + ((v.u >> 16) & 1u);   // RNE to bf16
    return (unsigned short)(r >> 16);
}
static __device__ __forceinline__ float b2f(unsigned short h) {
    unsigned u = ((unsigned)h) << 16;
    return __builtin_bit_cast(float, u);
}
static __device__ __forceinline__ __bf16 bcast(unsigned short h) {
    return __builtin_bit_cast(__bf16, h);
}
static __device__ __forceinline__ float sigf(float x) { return 1.0f / (1.0f + expf(-x)); }

// padded activation geometry
#define H1_ROW   34                 // 32 + halo
#define H1_FRM   (34 * 34 * 16)    // 18496 ushorts per frame
#define H2_ROW   18                 // 16 + halo
#define H2_FRM   (18 * 18 * 32)    // 10368 ushorts per frame

// ---------- weight packing ----------
// conv OIHW fp32 -> bf16 [O][Kpad], K = tap*Cin + c (Cin = 1<<cshift), zero padded
__global__ void pack_conv_w(const float* __restrict__ w, unsigned short* __restrict__ dst,
                            int O, int cshift, int Kpad) {
    int idx = blockIdx.x * 256 + threadIdx.x;
    if (idx >= O * Kpad) return;
    int o = idx / Kpad, K = idx % Kpad;
    int Cin = 1 << cshift;
    unsigned short v = 0;
    int tap = K >> cshift, c = K & (Cin - 1);
    if (tap < 9) v = f2b(w[(o * Cin + c) * 9 + tap]);
    dst[idx] = v;
}
// LSTM W fp32 [fan][512] -> bf16 transposed [512][fan]
__global__ void pack_lstm_w(const float* __restrict__ w, unsigned short* __restrict__ dst, int fan) {
    int idx = blockIdx.x * 256 + threadIdx.x;
    if (idx >= 512 * fan) return;
    int n = idx / fan, k = idx % fan;
    dst[idx] = f2b(w[k * 512 + n]);
}
__global__ void zero_f32(float* __restrict__ p, int n) {
    int idx = blockIdx.x * 256 + threadIdx.x;
    if (idx < n) p[idx] = 0.0f;
}

// ---------- stage 1: conv1 (1->16) + relu + maxpool2, fp32 VALU ----------
// out: interior of padded NHWC h1p [512][34][34][16]
__global__ void conv1_pool(const float* __restrict__ x, const float* __restrict__ w,
                           const float* __restrict__ b, unsigned short* __restrict__ h1p) {
    int idx = blockIdx.x * 256 + threadIdx.x;
    if (idx >= 512 * 1024 * 16) return;
    int ch = idx & 15, pos = (idx >> 4) & 1023, f = idx >> 14;
    int xo = pos & 31, yo = pos >> 5;
    const float* xf = x + (size_t)f * 4096;
    float m = -3.4e38f;
    #pragma unroll
    for (int py = 0; py < 2; ++py)
    #pragma unroll
    for (int px = 0; px < 2; ++px) {
        int cy = yo * 2 + py, cx = xo * 2 + px;
        float s = b[ch];
        #pragma unroll
        for (int t = 0; t < 9; ++t) {
            int iy = cy + t / 3 - 1, ix = cx + t % 3 - 1;
            if ((unsigned)iy < 64u && (unsigned)ix < 64u)
                s += xf[iy * 64 + ix] * w[ch * 9 + t];
        }
        m = fmaxf(m, fmaxf(s, 0.0f));
    }
    h1p[(size_t)f * H1_FRM + (((yo + 1) * H1_ROW) + xo + 1) * 16 + ch] = f2b(m);
}

// ---------- stage 2: conv2 (16->32) implicit-GEMM WMMA, relu ----------
// A: halo-padded NHWC h1p (no bounds checks); K = tap*16 + c, Kpad=160.
// out c2out bf16 NHWC [512][32][32][32]
__global__ void conv2_wmma(const unsigned short* __restrict__ h1p,
                           const unsigned short* __restrict__ Wc2,
                           const float* __restrict__ bias,
                           unsigned short* __restrict__ out) {
    int wave = threadIdx.x >> 5, lane = threadIdx.x & 31;
    int f = blockIdx.x >> 5;
    int tile = (blockIdx.x & 31) * 4 + wave;   // 0..127
    int mtile = tile >> 1, ntile = tile & 1;
    int hf = lane >> 4, l16 = lane & 15;
    int pos = mtile * 16 + l16;                // A row (position)
    int oy = pos >> 5, ox = pos & 31;
    int och = ntile * 16 + l16;                // B col (out channel)
    int coff = 8 * hf;
    // lane-invariant-per-tap offsets fold into load IOFFSET
    const unsigned short* abase = h1p + (size_t)f * H1_FRM
                                + (((oy + 1) * H1_ROW) + ox + 1) * 16 + coff;
    const unsigned short* wr = Wc2 + och * 160 + coff;
    v8f acc = {};
    #pragma unroll
    for (int kc = 0; kc < 5; ++kc) {           // fully unrolled: taps constant-fold
        const int t0 = 2 * kc;
        const int t1 = (2 * kc + 1 < 9) ? 2 * kc + 1 : 8;   // tap 9 -> any valid addr (B rows are 0)
        const int o0 = ((t0 / 3 - 1) * H1_ROW + (t0 % 3 - 1)) * 16;
        const int o1 = ((t1 / 3 - 1) * H1_ROW + (t1 % 3 - 1)) * 16;
        us8 v0 = *(const us8*)(abase + o0);
        us8 v1 = *(const us8*)(abase + o1);
        us8 b0 = *(const us8*)(wr + kc * 32);
        us8 b1 = *(const us8*)(wr + kc * 32 + 16);
        vbf16 a, bm;
        #pragma unroll
        for (int e = 0; e < 8; ++e) {
            a[e] = bcast(v0[e]);  a[e + 8] = bcast(v1[e]);
            bm[e] = bcast(b0[e]); bm[e + 8] = bcast(b1[e]);
        }
        acc = __builtin_amdgcn_wmma_f32_16x16x32_bf16(false, a, false, bm, (short)0, acc, false, false);
    }
    float bv = bias[och];
    #pragma unroll
    for (int r = 0; r < 8; ++r) {
        int p = mtile * 16 + r + 8 * hf;       // D: M = r + 8*half
        float v = fmaxf(acc[r] + bv, 0.0f);
        out[(((size_t)f * 1024) + p) * 32 + och] = f2b(v);
    }
}

// maxpool 2x2 on NHWC c2out -> interior of padded h2p [512][18][18][32]
__global__ void pool2(const unsigned short* __restrict__ src, unsigned short* __restrict__ h2p) {
    int idx = blockIdx.x * 256 + threadIdx.x;
    if (idx >= 512 * 256 * 32) return;
    int c = idx & 31, x = (idx >> 5) & 15, y = (idx >> 9) & 15, f = idx >> 13;
    const unsigned short* s = src + (((size_t)f * 1024) + (y * 2) * 32 + x * 2) * 32 + c;
    float m = fmaxf(fmaxf(b2f(s[0]), b2f(s[32])), fmaxf(b2f(s[1024]), b2f(s[1056])));
    h2p[(size_t)f * H2_FRM + (((y + 1) * H2_ROW) + x + 1) * 32 + c] = f2b(m);
}

// ---------- stage 3: conv3 (32->256) WMMA + relu + fused global-avg partial sums ----------
// K = tap*32 + c, Kdim = 288 (9 chunks, one tap per chunk), halo-padded input
__global__ void conv3_wmma(const unsigned short* __restrict__ h2p,
                           const unsigned short* __restrict__ Wc3,
                           const float* __restrict__ bias,
                           float* __restrict__ embsum) {
    int wave = threadIdx.x >> 5, lane = threadIdx.x & 31;
    int f = blockIdx.x >> 6;
    int tile = (blockIdx.x & 63) * 4 + wave;   // 0..255
    int mtile = tile >> 4, ntile = tile & 15;
    int hf = lane >> 4, l16 = lane & 15;
    int pos = mtile * 16 + l16;
    int oy = pos >> 4, ox = pos & 15;
    int och = ntile * 16 + l16;
    int coff = 8 * hf;
    const unsigned short* abase = h2p + (size_t)f * H2_FRM
                                + (((oy + 1) * H2_ROW) + ox + 1) * 32 + coff;
    const unsigned short* wr = Wc3 + och * 288 + coff;
    v8f acc = {};
    #pragma unroll
    for (int kc = 0; kc < 9; ++kc) {           // one tap per 32-wide chunk
        const int toff = ((kc / 3 - 1) * H2_ROW + (kc % 3 - 1)) * 32;  // constant
        us8 v0 = *(const us8*)(abase + toff);           // channels coff..coff+7
        us8 v1 = *(const us8*)(abase + toff + 16);      // channels 16+coff..
        us8 b0 = *(const us8*)(wr + kc * 32);
        us8 b1 = *(const us8*)(wr + kc * 32 + 16);
        vbf16 a, bm;
        #pragma unroll
        for (int e = 0; e < 8; ++e) {
            a[e] = bcast(v0[e]);  a[e + 8] = bcast(v1[e]);
            bm[e] = bcast(b0[e]); bm[e + 8] = bcast(b1[e]);
        }
        acc = __builtin_amdgcn_wmma_f32_16x16x32_bf16(false, a, false, bm, (short)0, acc, false, false);
    }
    float bv = bias[och], s = 0.0f;
    #pragma unroll
    for (int r = 0; r < 8; ++r) s += fmaxf(acc[r] + bv, 0.0f);
    atomicAdd(&embsum[f * 256 + och], s);      // lane's channel fixed -> one atomic per lane
}

__global__ void emb_finalize(const float* __restrict__ embsum, unsigned short* __restrict__ emb_bf) {
    int idx = blockIdx.x * 256 + threadIdx.x;
    if (idx >= 512 * 256) return;
    emb_bf[idx] = f2b(embsum[idx] * (1.0f / 256.0f));
}

// ---------- LSTM ----------
// comb0[b][0..255] = emb(b,t,:), [256..767] = bf16(h0)
__global__ void build_comb0(const unsigned short* __restrict__ emb_bf, const float* __restrict__ h0,
                            unsigned short* __restrict__ comb0, int t) {
    int idx = blockIdx.x * 256 + threadIdx.x;
    if (idx >= 32 * 768) return;
    int b = idx / 768, k = idx % 768;
    unsigned short v = (k < 256) ? emb_bf[(b * 16 + t) * 256 + k]
                                 : f2b(h0[b * 512 + (k - 256)]);
    comb0[idx] = v;
}

// GEMM: gates[g][32][512] = comb[32][K] @ Wt[g][512][K]^T  (Wt pre-transposed)
__global__ void lstm_gemm(const unsigned short* __restrict__ comb,
                          const unsigned short* __restrict__ Wt,
                          float* __restrict__ gates, int K) {
    int wave = threadIdx.x >> 5, lane = threadIdx.x & 31;
    int tile = blockIdx.x * 4 + wave;          // 0..255
    int mtile = tile >> 7, g = (tile >> 5) & 3, ntile = tile & 31;
    int hf = lane >> 4, l16 = lane & 15;
    int row = mtile * 16 + l16;
    int ncol = ntile * 16 + l16;
    const unsigned short* arow = comb + (size_t)row * K + 8 * hf;
    const unsigned short* brow = Wt + ((size_t)(g * 512 + ncol)) * K + 8 * hf;
    v8f acc = {};
    for (int kb = 0; kb < K; kb += 32) {
        __builtin_prefetch(brow + kb + 128, 0, 0);       // global_prefetch_b8, next chunks
        us8 a0 = *(const us8*)(arow + kb);
        us8 a1 = *(const us8*)(arow + kb + 16);
        us8 b0 = *(const us8*)(brow + kb);
        us8 b1 = *(const us8*)(brow + kb + 16);
        vbf16 a, bm;
        #pragma unroll
        for (int e = 0; e < 8; ++e) {
            a[e] = bcast(a0[e]); a[e + 8] = bcast(a1[e]);
            bm[e] = bcast(b0[e]); bm[e + 8] = bcast(b1[e]);
        }
        acc = __builtin_amdgcn_wmma_f32_16x16x32_bf16(false, a, false, bm, (short)0, acc, false, false);
    }
    #pragma unroll
    for (int r = 0; r < 8; ++r) {
        int M = mtile * 16 + r + 8 * hf;
        gates[(g * 32 + M) * 512 + ncol] = acc[r];
    }
}

// pointwise gate math; optionally builds comb for the next layer
__global__ void lstm_point(const float* __restrict__ gates,
                           const float* __restrict__ bf_, const float* __restrict__ bi_,
                           const float* __restrict__ bg_, const float* __restrict__ bo_,
                           float* __restrict__ cst, float* __restrict__ hst,
                           const float* __restrict__ h_other, unsigned short* __restrict__ comb_next) {
    int idx = blockIdx.x * 256 + threadIdx.x;
    if (idx >= 32 * 512) return;
    int b = idx >> 9, u = idx & 511;
    float fg = sigf(gates[(0 * 32 + b) * 512 + u] + bf_[u]);
    float ig = sigf(gates[(1 * 32 + b) * 512 + u] + bi_[u]);
    float gg = tanhf(gates[(2 * 32 + b) * 512 + u] + bg_[u]);
    float og = sigf(gates[(3 * 32 + b) * 512 + u] + bo_[u]);
    float c = fg * cst[idx] + ig * gg;
    float h = og * tanhf(c);
    cst[idx] = c; hst[idx] = h;
    if (comb_next) {
        comb_next[b * 1024 + u]       = f2b(h);
        comb_next[b * 1024 + 512 + u] = f2b(h_other[idx]);
    }
}

// ---------- classifier ----------
__global__ void fc_kernel(const float* __restrict__ in, const float* __restrict__ w,
                          const float* __restrict__ b, float* __restrict__ out,
                          int K, int N, int relu) {
    int idx = blockIdx.x * 256 + threadIdx.x;
    if (idx >= 32 * N) return;
    int bi = idx / N, n = idx % N;
    float s = b[n];
    for (int k = 0; k < K; ++k) s += in[bi * K + k] * w[k * N + n];
    if (relu) s = fmaxf(s, 0.0f);
    out[bi * N + n] = s;
}

// ---------- host ----------
extern "C" void kernel_launch(void* const* d_in, const int* in_sizes, int n_in,
                              void* d_out, int out_size, void* d_ws, size_t ws_size,
                              hipStream_t stream) {
    const float* x   = (const float*)d_in[0];
    const float* c1w = (const float*)d_in[1];  const float* c1b = (const float*)d_in[2];
    const float* c2w = (const float*)d_in[3];  const float* c2b = (const float*)d_in[4];
    const float* c3w = (const float*)d_in[5];  const float* c3b = (const float*)d_in[6];
    const float* cw1 = (const float*)d_in[23]; const float* cb1 = (const float*)d_in[24];
    const float* cw2 = (const float*)d_in[25]; const float* cb2 = (const float*)d_in[26];
    const float* cw3 = (const float*)d_in[27]; const float* cb3 = (const float*)d_in[28];

    char* base = (char*)d_ws;
    size_t off = 0;
    auto alloc = [&](size_t bytes) { void* p = base + off; off = (off + bytes + 255) & ~(size_t)255; return p; };

    unsigned short* h1p    = (unsigned short*)alloc((size_t)512 * H1_FRM * 2);  // padded NHWC
    unsigned short* c2out  = (unsigned short*)alloc((size_t)512 * 32768 * 2);   // NHWC
    unsigned short* h2p    = (unsigned short*)alloc((size_t)512 * H2_FRM * 2);  // padded NHWC
    float*          embsum = (float*)         alloc((size_t)512 * 256 * 4);
    unsigned short* emb_bf = (unsigned short*)alloc((size_t)512 * 256 * 2);
    unsigned short* Wc2    = (unsigned short*)alloc((size_t)32 * 160 * 2);
    unsigned short* Wc3    = (unsigned short*)alloc((size_t)256 * 288 * 2);
    unsigned short* Wt0    = (unsigned short*)alloc((size_t)4 * 512 * 768 * 2);
    unsigned short* Wt1    = (unsigned short*)alloc((size_t)4 * 512 * 1024 * 2);
    unsigned short* comb0  = (unsigned short*)alloc((size_t)32 * 768 * 2);
    unsigned short* comb1  = (unsigned short*)alloc((size_t)32 * 1024 * 2);
    float*          gates  = (float*)         alloc((size_t)4 * 32 * 512 * 4);
    float*          hstate = (float*)         alloc((size_t)2 * 32 * 512 * 4);
    float*          cstate = (float*)         alloc((size_t)2 * 32 * 512 * 4);
    float*          fc1    = (float*)         alloc((size_t)32 * 128 * 4);
    float*          fc2    = (float*)         alloc((size_t)32 * 64 * 4);
    (void)ws_size; (void)in_sizes; (void)n_in; (void)out_size;

    // zero padded activation buffers (halo must be 0 every call)
    {
        int n1 = 512 * H1_FRM / 2;   // ushorts -> floats
        int n2 = 512 * H2_FRM / 2;
        zero_f32<<<(n1 + 255) / 256, 256, 0, stream>>>((float*)h1p, n1);
        zero_f32<<<(n2 + 255) / 256, 256, 0, stream>>>((float*)h2p, n2);
    }
    // weight packing + state init
    pack_conv_w<<<(32 * 160 + 255) / 256, 256, 0, stream>>>(c2w, Wc2, 32, 4, 160);
    pack_conv_w<<<(256 * 288 + 255) / 256, 256, 0, stream>>>(c3w, Wc3, 256, 5, 288);
    for (int g = 0; g < 4; ++g) {
        pack_lstm_w<<<(512 * 768 + 255) / 256, 256, 0, stream>>>(
            (const float*)d_in[7 + 2 * g],  Wt0 + (size_t)g * 512 * 768, 768);
        pack_lstm_w<<<(512 * 1024 + 255) / 256, 256, 0, stream>>>(
            (const float*)d_in[15 + 2 * g], Wt1 + (size_t)g * 512 * 1024, 1024);
    }
    zero_f32<<<(512 * 256 + 255) / 256, 256, 0, stream>>>(embsum, 512 * 256);
    zero_f32<<<(2 * 32 * 512 + 255) / 256, 256, 0, stream>>>(hstate, 2 * 32 * 512);
    zero_f32<<<(2 * 32 * 512 + 255) / 256, 256, 0, stream>>>(cstate, 2 * 32 * 512);

    // encoder
    conv1_pool<<<(512 * 16384 + 255) / 256, 256, 0, stream>>>(x, c1w, c1b, h1p);
    conv2_wmma<<<512 * 32, 128, 0, stream>>>(h1p, Wc2, c2b, c2out);
    pool2<<<(512 * 8192 + 255) / 256, 256, 0, stream>>>(c2out, h2p);
    conv3_wmma<<<512 * 64, 128, 0, stream>>>(h2p, Wc3, c3b, embsum);
    emb_finalize<<<(512 * 256 + 255) / 256, 256, 0, stream>>>(embsum, emb_bf);

    // LSTM scan
    const float* bf0 = (const float*)d_in[8];  const float* bi0 = (const float*)d_in[10];
    const float* bg0 = (const float*)d_in[12]; const float* bo0 = (const float*)d_in[14];
    const float* bf1 = (const float*)d_in[16]; const float* bi1 = (const float*)d_in[18];
    const float* bg1 = (const float*)d_in[20]; const float* bo1 = (const float*)d_in[22];
    for (int t = 0; t < 16; ++t) {
        build_comb0<<<(32 * 768 + 255) / 256, 256, 0, stream>>>(emb_bf, hstate, comb0, t);
        lstm_gemm<<<64, 128, 0, stream>>>(comb0, Wt0, gates, 768);
        lstm_point<<<64, 256, 0, stream>>>(gates, bf0, bi0, bg0, bo0,
                                           cstate, hstate, hstate + 32 * 512, comb1);
        lstm_gemm<<<64, 128, 0, stream>>>(comb1, Wt1, gates, 1024);
        lstm_point<<<64, 256, 0, stream>>>(gates, bf1, bi1, bg1, bo1,
                                           cstate + 32 * 512, hstate + 32 * 512,
                                           (const float*)nullptr, (unsigned short*)nullptr);
    }

    // classifier head
    fc_kernel<<<(32 * 128 + 255) / 256, 256, 0, stream>>>(hstate + 32 * 512, cw1, cb1, fc1, 512, 128, 1);
    fc_kernel<<<(32 * 64 + 255) / 256, 256, 0, stream>>>(fc1, cw2, cb2, fc2, 128, 64, 1);
    fc_kernel<<<(32 * 6 + 255) / 256, 256, 0, stream>>>(fc2, cw3, cb3, (float*)d_out, 64, 6, 0);
}
